// MyModel_34720515621233
// MI455X (gfx1250) — matrix operand order repaired and checked by reading, not compiled, not run on hardware
//
#include <hip/hip_runtime.h>
#include <hip/hip_bf16.h>
#include <hip/hip_fp16.h>

typedef __attribute__((ext_vector_type(16))) _Float16 v16h;
typedef __attribute__((ext_vector_type(8)))  float    v8f;

#define B_   256
#define S_   200
#define E_   64
#define H_   128
#define N_   5000
#define NPAD 5008
#define HTLD 5024
#define TPAD 176
#define FIN  592

// ---------------- WMMA helpers (CDNA5 wave32, 16x16x32 f16 -> f32) ----------------

__device__ __forceinline__ v8f wmma16(v16h a, v16h b, v8f c) {
    return __builtin_amdgcn_wmma_f32_16x16x32_f16(false, a, false, b, (short)0, c, false, false);
}

// A fragment: 16x32 f16, row = M (lane&15), K interleaved per ISA table:
//   k(e) = (e&7) + (e>>3)*16 + (lane>>4)*8
__device__ __forceinline__ v16h load_a_frag(const _Float16* M, int ld, int row, int kc) {
    int lane = threadIdx.x & 31;
    int kh = lane >> 4;
    const _Float16* p = M + row * ld + kc + kh * 8;
    v16h f;
#pragma unroll
    for (int e = 0; e < 8; ++e) f[e] = p[e];
#pragma unroll
    for (int e = 0; e < 8; ++e) f[e + 8] = p[e + 16];
    return f;
}

// B fragment: 32x16 f16, K interleaved like A: k(e) = (e&7) + (e>>3)*16 + (lane>>4)*8.
// Source matrix must be N-major: row index = N, columns = K.
__device__ __forceinline__ v16h load_b_frag(const _Float16* M, int ld, int row, int kc) {
    int lane = threadIdx.x & 31;
    int kh = lane >> 4;
    const _Float16* p = M + row * ld + kc + kh * 8;
    v16h f;
#pragma unroll
    for (int e = 0; e < 8; ++e) f[e] = p[e];
#pragma unroll
    for (int e = 0; e < 8; ++e) f[e + 8] = p[e + 16];
    return f;
}

// float <-> order-preserving uint (for exact atomic max of signed floats)
__device__ __forceinline__ unsigned encf(float v) {
    unsigned u = __float_as_uint(v);
    return (u & 0x80000000u) ? ~u : (u | 0x80000000u);
}
__device__ __forceinline__ float decf(unsigned u) {
    return (u & 0x80000000u) ? __uint_as_float(u & 0x7fffffffu) : __uint_as_float(~u);
}

// ---------------- conversion / prep kernels ----------------

__global__ void k_cvt(const float* src, _Float16* dst, int n) {
    for (int i = blockIdx.x * blockDim.x + threadIdx.x; i < n; i += gridDim.x * blockDim.x)
        dst[i] = (_Float16)src[i];
}

__global__ void k_cvt_pad(const float* src, _Float16* dst, int valid_rows, int total_rows, int cols) {
    int n = total_rows * cols;
    for (int i = blockIdx.x * blockDim.x + threadIdx.x; i < n; i += gridDim.x * blockDim.x) {
        int r = i / cols;
        dst[i] = (r < valid_rows) ? (_Float16)src[i] : (_Float16)0.f;
    }
}

// transpose gat weights to N-major f16: wt[l][n][k] = W_l[k][n]
__global__ void k_wt(const float* gat_W, const float* gat_Wout, _Float16* wt) {
    int i = blockIdx.x * blockDim.x + threadIdx.x;
    if (i >= 4 * 64 * 64) return;
    int l = i >> 12, rem = i & 4095, n = rem >> 6, k = rem & 63;
    const float* srcm = (l < 3) ? (gat_W + l * 4096) : gat_Wout;
    wt[l * 4096 + n * 64 + k] = (_Float16)srcm[k * 64 + n];
}

__global__ void k_zero_tail(_Float16* hT) {
    int i = blockIdx.x * blockDim.x + threadIdx.x;
    if (i >= 64 * 16) return;
    int r = i >> 4, c = NPAD + (i & 15);
    hT[r * HTLD + c] = (_Float16)0.f;
}

__global__ void k_ts16(const float* tsum, _Float16* ts16) {
    int i = blockIdx.x * blockDim.x + threadIdx.x;
    if (i >= NPAD * 64) return;
    int r = i >> 6;
    ts16[i] = (r < N_) ? (_Float16)(tsum[i] * (1.f / 3.f)) : (_Float16)0.f;
}

// ---------------- cosine-similarity + masked max branch ----------------

__global__ void k_cos(const float* loc_emb, const int* loc1, const int* loc2,
                      const int* len1p, const int* len2p, float* feat) {
    __shared__ _Float16 L1[208 * 64];
    __shared__ _Float16 L2[208 * 64];
    __shared__ unsigned rmax[208];
    __shared__ unsigned cmax[208];
    int b = blockIdx.x;
    int t = threadIdx.x;
    int len1 = len1p[b], len2 = len2p[b];

    if (t < 208) {
        for (int which = 0; which < 2; ++which) {
            const int* loc = which ? loc2 : loc1;
            _Float16* L = which ? L2 : L1;
            if (t < S_) {
                int idx = loc[b * S_ + t];
                const float* src = loc_emb + (long)idx * E_;
                float ss = 0.f;
                for (int k = 0; k < E_; ++k) { float v = src[k]; ss += v * v; }
                float rn = rsqrtf(ss);
                for (int k = 0; k < E_; ++k) L[t * E_ + k] = (_Float16)(src[k] * rn);
            } else {
                for (int k = 0; k < E_; ++k) L[t * E_ + k] = (_Float16)0.f;
            }
        }
        rmax[t] = 0u; cmax[t] = 0u;
    }
    __syncthreads();

    int wave = t >> 5, lane = t & 31;
    for (int tile = wave; tile < 169; tile += 8) {
        int mt = tile / 13, nt = tile % 13;
        v8f acc = {};
        for (int kc = 0; kc < 64; kc += 32) {
            v16h a  = load_a_frag(L1, 64, mt * 16 + (lane & 15), kc);
            v16h bf = load_b_frag(L2, 64, nt * 16 + (lane & 15), kc);
            acc = wmma16(a, bf, acc);
        }
        int n = nt * 16 + (lane & 15);
        int mbase = mt * 16 + ((lane >> 4) << 3);
        // row maxes: predicate (n < len2) is the same for all 8 C registers
        if (n < len2) {
#pragma unroll
            for (int r = 0; r < 8; ++r)
                atomicMax(&rmax[mbase + r], encf(acc[r]));
        }
        // col max: pre-reduce the 8 rows (masked by m < len1) -> single atomic
        float cm = -3.0e38f;
#pragma unroll
        for (int r = 0; r < 8; ++r)
            cm = (mbase + r < len1) ? fmaxf(cm, acc[r]) : cm;
        if (cm > -2.9e38f) atomicMax(&cmax[n], encf(cm));
    }
    __syncthreads();
    if (t < S_) {
        feat[b * FIN + t]      = (t < len1) ? decf(rmax[t]) : 0.f;
        feat[b * FIN + S_ + t] = (t < len2) ? decf(cmax[t]) : 0.f;
    }
}

// ---------------- LSTM branch ----------------

// xg_table[t][n] = time_emb[t] . Wih[n] + bih[n] + bhh[n]   (176x512, K=64)
__global__ void k_xg(const _Float16* temb16, const _Float16* wih16,
                     const float* bih, const float* bhh, float* xg) {
    int gw = (blockIdx.x * blockDim.x + threadIdx.x) >> 5;  // 0..351
    int lane = threadIdx.x & 31;
    int mt = gw >> 5, nt = gw & 31;
    v8f acc = {};
    for (int kc = 0; kc < 64; kc += 32) {
        v16h a = load_a_frag(temb16, 64, mt * 16 + (lane & 15), kc);
        v16h b = load_b_frag(wih16, 64, nt * 16 + (lane & 15), kc);
        acc = wmma16(a, b, acc);
    }
    int n = nt * 16 + (lane & 15);
    float bias = bih[n] + bhh[n];
#pragma unroll
    for (int r = 0; r < 8; ++r) {
        int m = mt * 16 + r + ((lane >> 4) << 3);
        xg[m * 512 + n] = acc[r] + bias;
    }
}

// recurrent scan: 32 blocks = {seq 0/1} x {16 batch-blocks of 16}
// LDS: Whh f16 [512x128] (B-operand, N-major), gate f32 [16x512], hA f16 [16x128]
__global__ void k_lstm(const _Float16* whh16, const float* xg,
                       const int* time1, const int* time2,
                       const int* len1p, const int* len2p,
                       const float* time_v, float* p1ws, float* seqws) {
    extern __shared__ char smem[];
    _Float16* WH   = (_Float16*)smem;                              // 131072 B
    float*    gate = (float*)(smem + 131072);                      //  32768 B
    _Float16* hA   = (_Float16*)(smem + 131072 + 32768);           //   4096 B
    float*    pp   = (float*)(smem + 131072 + 32768 + 4096);       //   1024 B
    int*      tIdx = (int*)(smem + 131072 + 32768 + 4096 + 1024);  //     64 B
    int*      lenA = tIdx + 16;                                    //     64 B

    int t = threadIdx.x;
    int seq = blockIdx.x >> 4;
    int bb  = blockIdx.x & 15;
    const int* timep = seq ? time2 : time1;
    const int* lenp  = seq ? len2p : len1p;

    // async bulk stage: Whh f16 (128 KB) global -> LDS, no VGPR roundtrip
    {
        unsigned ldsBase = (unsigned)(unsigned long long)(void*)WH;
        const char* gsrc = (const char*)whh16;
        for (int i = t * 16; i < 512 * 128 * 2; i += 256 * 16) {
            unsigned ldst = ldsBase + (unsigned)i;
            const void* ga = gsrc + i;
            asm volatile("global_load_async_to_lds_b128 %0, %1, off"
                         :: "v"(ldst), "v"(ga) : "memory");
        }
        asm volatile("s_wait_asynccnt 0x0" ::: "memory");
    }
    for (int i = t; i < 16 * 128; i += 256) hA[i] = (_Float16)0.f;
    if (t < 16) lenA[t] = lenp[bb * 16 + t];
    __syncthreads();

    int wave = t >> 5, lane = t & 31;
    int lb = t >> 4;            // local batch for elementwise stage
    int hb = (t & 15) * 8;      // hidden base (8 per thread)
    float c[8];
    float tv[8];
#pragma unroll
    for (int j = 0; j < 8; ++j) { c[j] = 0.f; tv[j] = time_v[hb + j]; }

    for (int s = 0; s < S_; ++s) {
        if (t < 16) tIdx[t] = timep[(bb * 16 + t) * S_ + s];
        // gate = hA @ Whh^T : M=16, K=128, N=512 ; each wave owns 4 N-tiles.
        // A fragments are shared by all 4 N-tiles: load once.
        v16h aF[4];
#pragma unroll
        for (int kk = 0; kk < 4; ++kk)
            aF[kk] = load_a_frag(hA, 128, (lane & 15), kk * 32);
#pragma unroll
        for (int nt0 = 0; nt0 < 4; ++nt0) {
            int nt = wave * 4 + nt0;
            v8f acc = {};
#pragma unroll
            for (int kk = 0; kk < 4; ++kk) {
                v16h b = load_b_frag(WH, 128, nt * 16 + (lane & 15), kk * 32);
                acc = wmma16(aF[kk], b, acc);
            }
            int n = nt * 16 + (lane & 15);
#pragma unroll
            for (int r = 0; r < 8; ++r) {
                int m = r + ((lane >> 4) << 3);
                gate[m * 512 + n] = acc[r];
            }
        }
        __syncthreads();

        const float* xrow = xg + tIdx[lb] * 512;
        float ps = 0.f;
        int gb = bb * 16 + lb;
        int myLen = lenA[lb];
#pragma unroll
        for (int j = 0; j < 8; ++j) {
            int hh = hb + j;
            float iv = gate[lb * 512 + hh]       + xrow[hh];
            float fv = gate[lb * 512 + 128 + hh] + xrow[128 + hh];
            float gv = gate[lb * 512 + 256 + hh] + xrow[256 + hh];
            float ov = gate[lb * 512 + 384 + hh] + xrow[384 + hh];
            float ig = 1.f / (1.f + __expf(-iv));
            float fg = 1.f / (1.f + __expf(-fv));
            float og = 1.f / (1.f + __expf(-ov));
            c[j] = fg * c[j] + ig * tanhf(gv);
            float h = og * tanhf(c[j]);
            hA[lb * 128 + hh] = (_Float16)h;
            ps += h * tv[j];
            if (s == myLen - 1) seqws[(seq * B_ + gb) * H_ + hh] = h;
        }
        pp[lb * 16 + (t & 15)] = ps;
        __syncthreads();
        if (t < 16) {
            float sum = 0.f;
#pragma unroll
            for (int q = 0; q < 16; ++q) sum += pp[t * 16 + q];
            p1ws[(seq * B_ + bb * 16 + t) * S_ + s] = sum;
        }
        __syncthreads();
    }
}

__global__ void k_tloss(const float* p1ws, const float* tg1, const float* tg2,
                        const int* len1p, const int* len2p,
                        const float* wP, const float* bP, float* lossws) {
    __shared__ float red[256];
    int blk = blockIdx.x;           // 0..511
    int seq = blk >> 8, b = blk & 255;
    const float* tg = seq ? tg2 : tg1;
    int len = (seq ? len2p : len1p)[b];
    float w = wP[0], bb = bP[0];
    int t = threadIdx.x;
    float v = 0.f;
    if (t < S_ && t < len - 1) {
        float p1 = p1ws[(seq * B_ + b) * S_ + t];
        float tgv = (t + 1 < S_) ? tg[b * S_ + t + 1] : 0.f;
        float p2 = w * tgv;
        float p3 = __expf(p1 + bb) / w;
        float p4 = __expf(p1 + p2 + bb) / w;
        v = p1 + p2 + bb + p3 - p4;
    }
    red[t] = v;
    __syncthreads();
    for (int off = 128; off > 0; off >>= 1) {
        if (t < off) red[t] += red[t + off];
        __syncthreads();
    }
    if (t == 0) lossws[blk] = red[0] / (float)(len - 1);
}

__global__ void k_lfinal(const float* lossws, float* out) {
    if (threadIdx.x == 0 && blockIdx.x == 0) {
        float s = 0.f;
        for (int i = 0; i < 512; ++i) s += lossws[i];
        out[512] = -s / (float)B_;
    }
}

__global__ void k_tans(const float* seqws, float* feat) {
    int i = blockIdx.x * blockDim.x + threadIdx.x;
    if (i >= B_ * H_) return;
    int b = i >> 7, hh = i & 127;
    float v = seqws[b * H_ + hh] * seqws[(B_ + b) * H_ + hh];
    feat[b * FIN + 400 + hh] = tanhf(v);
}

// ---------------- GAT branch ----------------

// h = X @ W : [5008,64]x[64,64]; writes h row-major and N-major transpose
__global__ void k_gproj(const _Float16* X, const _Float16* Wt, _Float16* hr, _Float16* hT) {
    int gw = (blockIdx.x * blockDim.x + threadIdx.x) >> 5;
    int lane = threadIdx.x & 31;
    if (gw >= 313 * 4) return;          // wave-uniform exit
    int mt = gw >> 2, nt = gw & 3;
    v8f acc = {};
    for (int kc = 0; kc < 64; kc += 32) {
        v16h a = load_a_frag(X, 64, mt * 16 + (lane & 15), kc);
        v16h b = load_b_frag(Wt, 64, nt * 16 + (lane & 15), kc);
        acc = wmma16(a, b, acc);
    }
    int n = nt * 16 + (lane & 15);
#pragma unroll
    for (int r = 0; r < 8; ++r) {
        int m = mt * 16 + r + ((lane >> 4) << 3);
        _Float16 hv = (_Float16)acc[r];
        hr[m * 64 + n] = hv;
        hT[n * HTLD + m] = hv;
    }
}

__global__ void k_gf(const _Float16* hr, const float* a, float* f1, float* f2) {
    int i = blockIdx.x * blockDim.x + threadIdx.x;
    if (i >= NPAD) return;
    float s1 = 0.f, s2 = 0.f;
    for (int k = 0; k < 64; ++k) {
        float hv = (float)hr[i * 64 + k];
        s1 += hv * a[k];
        s2 += hv * a[64 + k];
    }
    f1[i] = s1; f2[i] = s2;
}

// attention aggregate: out[i] = elu( softmax_j(mask(lrelu(f1_i+f2_j))) @ h )
// rowmax exact via monotonicity: max_j lrelu(f1_i+f2_j) = lrelu(f1_i + max_j f2_j)
__global__ void k_gat(const float* adj, const float* f1, const float* f2,
                      const _Float16* hT, float* dst, int accumulate) {
    __shared__ float pp1[256];
    __shared__ float rm[16];
    __shared__ float cbuf[8][16][64];
    __shared__ float zbuf[8][32];
    __shared__ float Zm[16];
    int rb = blockIdx.x, i0 = rb * 16;
    int t = threadIdx.x;
    int il = t >> 4;
    int i = i0 + il;

    // pass 1: per-row max of f2 over adjacency (streaming, prefetch ahead)
    float m = -3.0e38f;
    if (i < N_) {
        for (int j = (t & 15); j < N_; j += 16) {
            if ((j >> 4 & 31) == 0)
                __builtin_prefetch(&adj[(long)i * N_ + j + 512], 0, 0);
            if (adj[(long)i * N_ + j] > 0.f) m = fmaxf(m, f2[j]);
        }
    }
    pp1[t] = m;
    __syncthreads();
    if (t < 16) {
        float mm = -3.0e38f;
        for (int q = 0; q < 16; ++q) mm = fmaxf(mm, pp1[t * 16 + q]);
        int ii = i0 + t;
        float e = (ii < N_) ? (f1[ii] + mm) : 0.f;
        rm[t] = (ii < N_) ? (e > 0.f ? e : 0.2f * e) : 0.f;
    }
    __syncthreads();

    // pass 2: P-tiles (16x32 f16) built on the fly, WMMA vs hT
    int wave = t >> 5, lane = t & 31;
    int ml = lane & 15, kh = lane >> 4;
    int irow = i0 + ml;
    float fi = (irow < N_) ? f1[irow] : 0.f;
    float rmax = rm[ml];
    v8f acc[4] = {};
    float zacc = 0.f;
    for (int cidx = wave; cidx < 157; cidx += 8) {
        int jbase = cidx * 32;
        if (irow < N_ && jbase + 256 < N_)
            __builtin_prefetch(&adj[(long)irow * N_ + jbase + 256], 0, 0);
        v16h a;
#pragma unroll
        for (int e = 0; e < 16; ++e) {
            int k = (e & 7) + ((e >> 3) << 4) + kh * 8;
            int j = jbase + k;
            float p = 0.f;
            if (irow < N_ && j < N_ && adj[(long)irow * N_ + j] > 0.f) {
                float ev = fi + f2[j];
                ev = ev > 0.f ? ev : 0.2f * ev;
                p = __expf(ev - rmax);
            }
            a[e] = (_Float16)p;
            zacc += p;
        }
#pragma unroll
        for (int nt = 0; nt < 4; ++nt) {
            v16h b = load_b_frag(hT, HTLD, nt * 16 + ml, jbase);
            acc[nt] = wmma16(a, b, acc[nt]);
        }
    }
#pragma unroll
    for (int nt = 0; nt < 4; ++nt)
#pragma unroll
        for (int r = 0; r < 8; ++r)
            cbuf[wave][r + (kh << 3)][nt * 16 + ml] = acc[nt][r];
    zbuf[wave][lane] = zacc;
    __syncthreads();
    if (t < 16) {
        float z = 0.f;
        for (int wq = 0; wq < 8; ++wq) z += zbuf[wq][t] + zbuf[wq][t + 16];
        Zm[t] = z;
    }
    __syncthreads();
    for (int e = t; e < 16 * 64; e += 256) {
        int mr = e >> 6, n = e & 63;
        int ii = i0 + mr;
        if (ii >= N_) continue;
        float sum = 0.f;
        for (int wq = 0; wq < 8; ++wq) sum += cbuf[wq][mr][n];
        float v = sum / Zm[mr];
        float o = v > 0.f ? v : (__expf(v) - 1.f);
        if (accumulate) dst[ii * 64 + n] += o;
        else            dst[ii * 64 + n]  = o;
    }
}

__global__ void k_had(const float* gat, const int* u1, const int* u2, float* feat) {
    int i = blockIdx.x * blockDim.x + threadIdx.x;
    if (i >= B_ * 64) return;
    int b = i >> 6, k = i & 63;
    float v = gat[u1[b] * 64 + k] * gat[u2[b] * 64 + k];
    feat[b * FIN + 528 + k] = tanhf(v);
}

__global__ void k_fc(const float* feat, const float* fcw, const float* fcb, float* out) {
    int b = threadIdx.x;
    float a0 = fcb[0], a1 = fcb[1];
    for (int j = 0; j < FIN; ++j) {
        float f = feat[b * FIN + j];
        a0 += f * fcw[j];
        a1 += f * fcw[FIN + j];
    }
    out[b * 2]     = a0;
    out[b * 2 + 1] = a1;
}

// ---------------- launch ----------------

extern "C" void kernel_launch(void* const* d_in, const int* in_sizes, int n_in,
                              void* d_out, int out_size, void* d_ws, size_t ws_size,
                              hipStream_t stream) {
    const int*   u1    = (const int*)d_in[0];
    const int*   u2    = (const int*)d_in[1];
    const int*   len1  = (const int*)d_in[2];
    const int*   len2  = (const int*)d_in[3];
    const int*   loc1  = (const int*)d_in[4];
    const int*   loc2  = (const int*)d_in[5];
    const int*   time1 = (const int*)d_in[6];
    const int*   time2 = (const int*)d_in[7];
    const float* tg1   = (const float*)d_in[8];
    const float* tg2   = (const float*)d_in[9];
    const float* loc_emb  = (const float*)d_in[10];
    const float* time_emb = (const float*)d_in[11];
    const float* user_emb = (const float*)d_in[12];
    const float* gat_W    = (const float*)d_in[13];
    const float* gat_a    = (const float*)d_in[14];
    const float* gat_Wout = (const float*)d_in[15];
    const float* gat_aout = (const float*)d_in[16];
    const float* adj      = (const float*)d_in[17];
    const float* Wih      = (const float*)d_in[18];
    const float* Whh      = (const float*)d_in[19];
    const float* bih      = (const float*)d_in[20];
    const float* bhh      = (const float*)d_in[21];
    const float* time_v   = (const float*)d_in[22];
    const float* time_w   = (const float*)d_in[23];
    const float* time_b   = (const float*)d_in[24];
    const float* fc_w     = (const float*)d_in[25];
    const float* fc_b     = (const float*)d_in[26];
    float* out = (float*)d_out;

    char* ws = (char*)d_ws;
    size_t off = 0;
    auto alloc = [&](size_t bytes) { char* p = ws + off; off += (bytes + 255) & ~(size_t)255; return p; };
    _Float16* wih16  = (_Float16*)alloc(512 * 64 * 2);
    _Float16* whh16  = (_Float16*)alloc(512 * 128 * 2);
    _Float16* temb16 = (_Float16*)alloc(TPAD * 64 * 2);
    _Float16* x16    = (_Float16*)alloc(NPAD * 64 * 2);
    _Float16* wt16   = (_Float16*)alloc(4 * 64 * 64 * 2);
    _Float16* hr16   = (_Float16*)alloc(NPAD * 64 * 2);
    _Float16* hT16   = (_Float16*)alloc(64 * HTLD * 2);
    _Float16* ts16   = (_Float16*)alloc(NPAD * 64 * 2);
    float* f1ws   = (float*)alloc(NPAD * 4);
    float* f2ws   = (float*)alloc(NPAD * 4);
    float* tsum   = (float*)alloc(NPAD * 64 * 4);
    float* gout   = (float*)alloc(NPAD * 64 * 4);
    float* xg     = (float*)alloc(TPAD * 512 * 4);
    float* p1ws   = (float*)alloc(2 * B_ * S_ * 4);
    float* seqws  = (float*)alloc(2 * B_ * H_ * 4);
    float* lossws = (float*)alloc(512 * 4);
    float* feat   = (float*)alloc(B_ * FIN * 4);
    (void)in_sizes; (void)n_in; (void)out_size; (void)ws_size;

    // prep / conversions
    k_cvt<<<128, 256, 0, stream>>>(Wih, wih16, 512 * 64);
    k_cvt<<<256, 256, 0, stream>>>(Whh, whh16, 512 * 128);
    k_cvt_pad<<<44, 256, 0, stream>>>(time_emb, temb16, 168, TPAD, 64);
    k_cvt_pad<<<1252, 256, 0, stream>>>(user_emb, x16, N_, NPAD, 64);
    k_wt<<<64, 256, 0, stream>>>(gat_W, gat_Wout, wt16);
    k_zero_tail<<<4, 256, 0, stream>>>(hT16);

    // cosine branch -> feat[:, 0:400]
    k_cos<<<256, 256, 0, stream>>>(loc_emb, loc1, loc2, len1, len2, feat);

    // LSTM branch
    k_xg<<<44, 256, 0, stream>>>(temb16, wih16, bih, bhh, xg);
    hipFuncSetAttribute(reinterpret_cast<const void*>(k_lstm),
                        hipFuncAttributeMaxDynamicSharedMemorySize, 169088);
    k_lstm<<<32, 256, 169088, stream>>>(whh16, xg, time1, time2, len1, len2,
                                        time_v, p1ws, seqws);
    k_tloss<<<512, 256, 0, stream>>>(p1ws, tg1, tg2, len1, len2, time_w, time_b, lossws);
    k_lfinal<<<1, 64, 0, stream>>>(lossws, out);
    k_tans<<<128, 256, 0, stream>>>(seqws, feat);

    // GAT branch: 3 heads accumulate into tsum, then output layer
    for (int l = 0; l < 3; ++l) {
        k_gproj<<<157, 256, 0, stream>>>(x16, wt16 + l * 4096, hr16, hT16);
        k_gf<<<20, 256, 0, stream>>>(hr16, gat_a + l * 128, f1ws, f2ws);
        k_gat<<<313, 256, 0, stream>>>(adj, f1ws, f2ws, hT16, tsum, (l > 0) ? 1 : 0);
    }
    k_ts16<<<1252, 256, 0, stream>>>(tsum, ts16);
    k_gproj<<<157, 256, 0, stream>>>(ts16, wt16 + 3 * 4096, hr16, hT16);
    k_gf<<<20, 256, 0, stream>>>(hr16, gat_aout, f1ws, f2ws);
    k_gat<<<313, 256, 0, stream>>>(adj, f1ws, f2ws, hT16, gout, 0);
    k_had<<<64, 256, 0, stream>>>(gout, u1, u2, feat);

    // final FC -> out[0:512]
    k_fc<<<1, 256, 0, stream>>>(feat, fc_w, fc_b, out);
}
